// IntrospectiveFlashDiff1_21234318311747
// MI455X (gfx1250) — matrix operand order repaired and checked by reading, not compiled, not run on hardware
//
#include <hip/hip_runtime.h>
#include <cstdint>

// ---------------- problem constants ----------------
#define BB 2
#define TT 1024
#define EE_DIM 2048
#define HH 16
#define DD 128
#define NL 3

typedef __bf16 bf16;
typedef __attribute__((ext_vector_type(8)))  bf16  bf16x8;
typedef __attribute__((ext_vector_type(16))) bf16  bf16x16;
typedef __attribute__((ext_vector_type(8)))  float f32x8;
typedef __attribute__((ext_vector_type(4)))  unsigned int u32x4;
typedef __attribute__((ext_vector_type(4)))  int i32x4;
typedef __attribute__((ext_vector_type(8)))  int i32x8;

// Tensor Data Mover availability (ROCm7.2 clang-22: 5-arg, amdgpu-toolchain clang-23: 6-arg)
#if defined(__has_builtin)
#  if __has_builtin(__builtin_amdgcn_tensor_load_to_lds) && __has_builtin(__builtin_amdgcn_s_wait_tensorcnt)
#    define HAVE_TDM 1
#  endif
#endif
#ifndef HAVE_TDM
#  define HAVE_TDM 0
#endif

static __device__ inline bf16x8 ld8(const bf16* p) { return *(const bf16x8*)p; }
static __device__ inline bf16x16 cat8(bf16x8 lo, bf16x8 hi) {
  return __builtin_shufflevector(lo, hi, 0,1,2,3,4,5,6,7,8,9,10,11,12,13,14,15);
}
static __device__ inline f32x8 zero8() {
  f32x8 z = {0.f,0.f,0.f,0.f,0.f,0.f,0.f,0.f};
  return z;
}
static __device__ inline f32x8 wmma_bf16(bf16x16 a, bf16x16 b, f32x8 c) {
  // D = A(16x32) * B(32x16) + C, f32 accumulate
  return __builtin_amdgcn_wmma_f32_16x16x32_bf16(false, a, false, b, (short)0, c, false, false);
}
static __device__ inline unsigned pack2(bf16 a, bf16 b) {
  union { bf16 h[2]; unsigned u; } p;
  p.h[0] = a; p.h[1] = b;
  return p.u;
}

#if HAVE_TDM
// Issue a TDM 2D tile load: tile_h rows x tile_w bf16 elements, row stride = row_stride
// elements, into contiguous LDS at lds_off. D# layout per CDNA5 ISA ch.8 (group0/group1).
static __device__ inline void tdm_load_2d(unsigned lds_off, const void* gptr,
                                          unsigned tile_w, unsigned tile_h,
                                          unsigned row_stride) {
  uint64_t ga = (uint64_t)(uintptr_t)gptr;
  u32x4 g0;
  g0[0] = 1u;                                             // count=1 (valid user D#)
  g0[1] = lds_off;                                        // lds_addr (bytes)
  g0[2] = (unsigned)(ga & 0xFFFFFFFFu);                   // global_addr[31:0]
  g0[3] = (unsigned)((ga >> 32) & 0x01FFFFFFu) | (2u << 30); // global_addr[56:32], type=2
  const unsigned td0 = row_stride;                        // tensor_dim0 (elements)
  const unsigned td1 = 1u << 20;                          // tensor_dim1: large (no OOB clip)
  i32x8 g1;
  g1[0] = (int)(1u << 16);                                // data_size=1 (2 bytes); wg_mask=0
  g1[1] = (int)((td0 & 0xFFFFu) << 16);                   // tensor_dim0[15:0]
  g1[2] = (int)(((td0 >> 16) & 0xFFFFu) | ((td1 & 0xFFFFu) << 16)); // dim0 hi | dim1 lo
  g1[3] = (int)(((td1 >> 16) & 0xFFFFu) | (tile_w << 16));          // dim1 hi | tile_dim0
  g1[4] = (int)(tile_h & 0xFFFFu);                        // tile_dim1 (tile_dim2=0)
  g1[5] = (int)td0;                                       // tensor_dim0_stride[31:0]
  g1[6] = 0;                                              // stride hi, dim1_stride lo
  g1[7] = 0;
  i32x4 gz = {0, 0, 0, 0};
#if __clang_major__ >= 23
  i32x8 gz8 = {0, 0, 0, 0, 0, 0, 0, 0};
  __builtin_amdgcn_tensor_load_to_lds(g0, g1, gz, gz, gz8, 0);
#else
  __builtin_amdgcn_tensor_load_to_lds(g0, g1, gz, gz, 0);
#endif
}
#endif

// ---------------- f32 -> bf16 cast ----------------
__global__ void cast_f32_bf16_kernel(const float* __restrict__ src, bf16* __restrict__ dst, size_t n) {
  size_t i = blockIdx.x * (size_t)blockDim.x + threadIdx.x;
  size_t stride = gridDim.x * (size_t)blockDim.x;
  for (; i < n; i += stride) dst[i] = (bf16)src[i];
}

// ---------------- tiled WMMA GEMM: C[M,N] = A[M,K] * B[N,K]^T ----------------
// A, B row-major bf16 (K contiguous) -> matches x @ W.T with W = [N,K].
// Double-buffered LDS tiles; TDM (tensor_load_to_lds) streams tiles when available.
template <typename OUT>
__global__ __launch_bounds__(256) void gemm_wmma_kernel(const bf16* __restrict__ A,
                                                        const bf16* __restrict__ Bw,
                                                        OUT* __restrict__ C,
                                                        int M, int N, int K) {
  __shared__ bf16 As[2][128 * 32];
  __shared__ bf16 Bs[2][128 * 32];
  const int tid  = threadIdx.x;
  const int lane = tid & 31;
  const int wave = tid >> 5;
  const int wm   = wave >> 2;  // 0..1
  const int wn   = wave & 3;   // 0..3
  const int m0   = blockIdx.y * 128;
  const int n0   = blockIdx.x * 128;
  const int half = lane >> 4;
  const int l15  = lane & 15;

  f32x8 acc[4][2];
  for (int m = 0; m < 4; ++m)
    for (int n = 0; n < 2; ++n) acc[m][n] = zero8();

#if HAVE_TDM
  const bool issuer = (wave == 0);
  if (issuer) {
    tdm_load_2d((unsigned)(uintptr_t)&As[0][0], &A [(size_t)m0 * K], 32u, 128u, (unsigned)K);
    tdm_load_2d((unsigned)(uintptr_t)&Bs[0][0], &Bw[(size_t)n0 * K], 32u, 128u, (unsigned)K);
  }
#endif

  int cur = 0;
  for (int kb = 0; kb < K; kb += 32) {
#if HAVE_TDM
    if (issuer) {
      if (kb + 32 < K) {
        // prefetch next tiles into the other buffer, then wait for the current pair
        tdm_load_2d((unsigned)(uintptr_t)&As[cur ^ 1][0], &A [(size_t)m0 * K + kb + 32], 32u, 128u, (unsigned)K);
        tdm_load_2d((unsigned)(uintptr_t)&Bs[cur ^ 1][0], &Bw[(size_t)n0 * K + kb + 32], 32u, 128u, (unsigned)K);
        __builtin_amdgcn_s_wait_tensorcnt((short)2);
      } else {
        __builtin_amdgcn_s_wait_tensorcnt((short)0);
      }
    }
#else
    // cooperative staging fallback (each thread: 2 x 16B per tile)
    for (int c = tid; c < 512; c += 256) {
      int row = c >> 2;
      int kc  = (c & 3) << 3;
      *(bf16x8*)&As[cur][row * 32 + kc] = ld8(&A [(size_t)(m0 + row) * K + kb + kc]);
      *(bf16x8*)&Bs[cur][row * 32 + kc] = ld8(&Bw[(size_t)(n0 + row) * K + kb + kc]);
    }
    if (kb + 32 < K) {
      __builtin_prefetch(&A [(size_t)(m0 + (tid >> 1)) * K + kb + 32], 0, 0);
      __builtin_prefetch(&Bw[(size_t)(n0 + (tid >> 1)) * K + kb + 32], 0, 0);
    }
#endif
    __syncthreads();

    bf16x16 bfrag[2];
    for (int n = 0; n < 2; ++n) {
      int nr = wn * 32 + n * 16 + l15;
      int k0 = half * 16;
      bfrag[n] = cat8(ld8(&Bs[cur][nr * 32 + k0]), ld8(&Bs[cur][nr * 32 + k0 + 8]));
    }
    for (int m = 0; m < 4; ++m) {
      int mr = wm * 64 + m * 16 + l15;
      int k0 = half * 16;
      bf16x16 afrag = cat8(ld8(&As[cur][mr * 32 + k0]), ld8(&As[cur][mr * 32 + k0 + 8]));
      for (int n = 0; n < 2; ++n) acc[m][n] = wmma_bf16(afrag, bfrag[n], acc[m][n]);
    }
    __syncthreads();
    cur ^= 1;
  }

  for (int m = 0; m < 4; ++m)
    for (int n = 0; n < 2; ++n) {
      union { f32x8 v; float s[8]; } u;
      u.v = acc[m][n];
      int col = n0 + wn * 32 + n * 16 + l15;
      int rb  = m0 + wm * 64 + m * 16 + (half << 3);
      for (int r = 0; r < 8; ++r)
        C[(size_t)(rb + r) * N + col] = (OUT)u.s[r];
    }
}

// ---------------- RoPE on Q (+ 1/sqrt(D) fold) ----------------
__global__ void rope_q_kernel(const bf16* __restrict__ q, const float* __restrict__ cosb,
                              const float* __restrict__ sinb, bf16* __restrict__ qo) {
  size_t idx = blockIdx.x * (size_t)blockDim.x + threadIdx.x;
  size_t tot = (size_t)BB * TT * HH * 64;
  if (idx >= tot) return;
  int d2 = idx & 63;
  size_t r = idx >> 6;
  int h = (int)(r & (HH - 1)); r >>= 4;
  int t = (int)(r % TT);
  int b = (int)(r / TT);
  size_t base = (((size_t)b * TT + t) * HH + h) * DD;
  float c = cosb[(size_t)t * 64 + d2];
  float s = sinb[(size_t)t * 64 + d2];
  float x1 = (float)q[base + d2];
  float x2 = (float)q[base + d2 + 64];
  const float scale = 0.08838834764831845f;  // 1/sqrt(128)
  qo[base + d2]      = (bf16)((x1 * c - x2 * s) * scale);
  qo[base + d2 + 64] = (bf16)((x1 * s + x2 * c) * scale);
}

// ---------------- build concatenated K (rope at concat position) and V^T ----------------
// kraw/vraw: NL consecutive [B,T,H,D] bf16 chunks, layer stride = B*T*E elems.
// Kc: [B,L,H,D]; Vt: [B,H,D,L]
__global__ void build_kv_kernel(const bf16* __restrict__ kraw, const bf16* __restrict__ vraw,
                                const float* __restrict__ cosb, const float* __restrict__ sinb,
                                bf16* __restrict__ Kc, bf16* __restrict__ Vt,
                                int L, int sg0, int sg1, int sg2, int sg3) {
  size_t idx = blockIdx.x * (size_t)blockDim.x + threadIdx.x;
  size_t tot = (size_t)BB * L * HH * 64;
  if (idx >= tot) return;
  int d2 = idx & 63;
  size_t r = idx >> 6;
  int h = (int)(r & (HH - 1)); r >>= 4;
  int s = (int)(r % L);
  int b = (int)(r / L);
  int sarr[4] = {sg0, sg1, sg2, sg3};
  int src  = sarr[s >> 10];           // T = 1024
  int srow = s & (TT - 1);
  const size_t LAYER = (size_t)BB * TT * EE_DIM;
  size_t sbase = src * LAYER + (((size_t)b * TT + srow) * HH + h) * DD;
  // K with rope at concat position s
  float c  = cosb[(size_t)s * 64 + d2];
  float sn = sinb[(size_t)s * 64 + d2];
  float k1 = (float)kraw[sbase + d2];
  float k2 = (float)kraw[sbase + d2 + 64];
  size_t kdst = (((size_t)b * L + s) * HH + h) * DD;
  Kc[kdst + d2]      = (bf16)(k1 * c - k2 * sn);
  Kc[kdst + d2 + 64] = (bf16)(k1 * sn + k2 * c);
  // V transposed
  size_t vbase = (((size_t)b * HH + h) * DD) * (size_t)L;
  Vt[vbase + (size_t)d2 * L + s]        = vraw[sbase + d2];
  Vt[vbase + (size_t)(d2 + 64) * L + s] = vraw[sbase + d2 + 64];
}

// ---------------- flash attention (bottom-right causal), WMMA ----------------
// Q: [B,T,H,D] roped+scaled bf16; Kc: [B,L,H,D]; Vt: [B,H,D,L]; O: [B,T,H,D] f32
// 4 waves/block share one (b,h); K/V tiles staged in LDS for 4x reuse.
// Block-uniform s-loop (waves mask blocks beyond their causal limit).
__global__ __launch_bounds__(128) void flash_attn_kernel(const bf16* __restrict__ Q,
                                                         const bf16* __restrict__ Kc,
                                                         const bf16* __restrict__ Vt,
                                                         float* __restrict__ O, int L) {
  __shared__ bf16 Ks[32 * 128];  // [s][d]
  __shared__ bf16 Vs[128 * 32];  // [d][s]
  const int tid  = threadIdx.x;
  const int lane = tid & 31;
  const int wave = tid >> 5;
  const int qblk = blockIdx.x * 64;
  const int qb   = qblk + wave * 16;
  const int bh   = blockIdx.y;
  const int b    = bh >> 4;   // H = 16
  const int h    = bh & 15;
  const int off  = L - TT;
  const int l15  = lane & 15;
  const int half = lane >> 4;
  const int tq   = qb + l15;  // this lane's q column (global q row)

  // Q^T B-fragments (N = q col, K = d), one per 32-wide d chunk
  bf16x16 qf[4];
  {
    const bf16* qrow = Q + (((size_t)b * TT + tq) * HH + h) * DD;
    for (int c = 0; c < 4; ++c) {
      int d0 = c * 32 + half * 16;
      qf[c] = cat8(ld8(qrow + d0), ld8(qrow + d0 + 8));
    }
  }

  float mstat = -1e30f, lstat = 0.f;
  f32x8 of[8];
  for (int f = 0; f < 8; ++f) of[f] = zero8();

  const int my_smax  = L < (qb + 16 + off) ? L : (qb + 16 + off);
  const int blk_smax = L < (qblk + 64 + off) ? L : (qblk + 64 + off);

  for (int sb = 0; sb < blk_smax; sb += 32) {
    // ---- cooperative staging of K (32 x 128) and V^T (128 x 32) tiles
    for (int c = tid; c < 512; c += 128) {
      int kr = c >> 4, kd = (c & 15) << 3;
      *(bf16x8*)&Ks[kr * 128 + kd] =
          ld8(&Kc[(((size_t)b * L + sb + kr) * HH + h) * DD + kd]);
      int vr = c >> 2, vs = (c & 3) << 3;
      *(bf16x8*)&Vs[vr * 32 + vs] =
          ld8(&Vt[(((size_t)b * HH + h) * DD + vr) * (size_t)L + sb + vs]);
    }
    if (sb + 32 < blk_smax)
      __builtin_prefetch(&Kc[(((size_t)b * L + sb + 32 + (tid >> 4)) * HH + h) * DD], 0, 0);
    __syncthreads();

    if (sb < my_smax) {
      // ---- S^T = K * Q^T : two 16x16 tiles (s rows in VGPRs, q cols in lanes)
      union { f32x8 v; float f[8]; } st[2];
      for (int tt = 0; tt < 2; ++tt) {
        f32x8 acc = zero8();
        int srow = tt * 16 + l15;  // local A-matrix row (M = s)
        for (int c = 0; c < 4; ++c) {
          int d0 = c * 32 + half * 16;
          bf16x16 kf = cat8(ld8(&Ks[srow * 128 + d0]), ld8(&Ks[srow * 128 + d0 + 8]));
          acc = wmma_bf16(kf, qf[c], acc);
        }
        st[tt].v = acc;
      }
      // ---- mask + online softmax (per q column == per lane; halves combined via xor16)
      float bm = -1e30f;
      for (int r = 0; r < 8; ++r) {
        int s0 = sb + half * 8 + r;
        if (s0 > tq + off)      st[0].f[r] = -1e30f;
        if (s0 + 16 > tq + off) st[1].f[r] = -1e30f;
        float a = st[0].f[r] > st[1].f[r] ? st[0].f[r] : st[1].f[r];
        bm = bm > a ? bm : a;
      }
      bm = fmaxf(bm, __shfl_xor(bm, 16));
      float mnew  = fmaxf(mstat, bm);
      float alpha = __expf(mstat - mnew);
      float ps = 0.f;
      for (int r = 0; r < 8; ++r) {
        st[0].f[r] = __expf(st[0].f[r] - mnew);
        st[1].f[r] = __expf(st[1].f[r] - mnew);
        ps += st[0].f[r] + st[1].f[r];
      }
      ps += __shfl_xor(ps, 16);
      lstat = lstat * alpha + ps;
      mstat = mnew;
      for (int f = 0; f < 8; ++f) of[f] = of[f] * alpha;

      // ---- build P^T B-fragment (K = s 0..31, N = q) via 2 x 4 dword xor16 exchanges
      unsigned t0[4], t1[4];
      for (int j = 0; j < 4; ++j) {
        t0[j] = pack2((bf16)st[0].f[2 * j], (bf16)st[0].f[2 * j + 1]);
        t1[j] = pack2((bf16)st[1].f[2 * j], (bf16)st[1].f[2 * j + 1]);
      }
      union { unsigned u[8]; bf16x16 v; } pf;
      for (int j = 0; j < 4; ++j) {
        unsigned x0 = (unsigned)__shfl_xor((int)t0[j], 16);
        unsigned x1 = (unsigned)__shfl_xor((int)t1[j], 16);
        if (half == 0) { pf.u[j] = t0[j]; pf.u[4 + j] = x0; }    // s 0..7  | 8..15
        else           { pf.u[j] = x1;    pf.u[4 + j] = t1[j]; } // s 16..23| 24..31
      }

      // ---- O^T += V^T * P^T  (8 d-blocks of 16)
      for (int f = 0; f < 8; ++f) {
        int drow = f * 16 + l15;
        int s0 = half * 16;
        bf16x16 vf = cat8(ld8(&Vs[drow * 32 + s0]), ld8(&Vs[drow * 32 + s0 + 8]));
        of[f] = wmma_bf16(vf, pf.v, of[f]);
      }
    }
    __syncthreads();
  }

  // ---- write O (lane: fixed q, 8 consecutive d per frag)
  float inv = 1.f / lstat;
  for (int f = 0; f < 8; ++f) {
    union { f32x8 v; float s[8]; } u;
    u.v = of[f];
    float* orow = O + (((size_t)b * TT + tq) * HH + h) * DD + f * 16 + half * 8;
    for (int r = 0; r < 8; ++r) orow[r] = u.s[r] * inv;
  }
}

// ---------------- RMS norm + lambda-weighted accumulate ----------------
__global__ __launch_bounds__(256) void rms_accum_kernel(const float* __restrict__ attn,
                                                        const float* __restrict__ ln_w,
                                                        const float* __restrict__ lambdas,
                                                        int layer, float* __restrict__ total,
                                                        int first) {
  int row = blockIdx.x;  // 0 .. B*T-1
  const float* a = attn + (size_t)row * EE_DIM;
  float ss = 0.f;
  for (int e = threadIdx.x; e < EE_DIM; e += 256) { float v = a[e]; ss += v * v; }
  __shared__ float red[256];
  red[threadIdx.x] = ss;
  __syncthreads();
  for (int w = 128; w > 0; w >>= 1) {
    if (threadIdx.x < w) red[threadIdx.x] += red[threadIdx.x + w];
    __syncthreads();
  }
  float inv = rsqrtf(red[0] / (float)EE_DIM + 1e-5f);
  // lambda weights: sigmoid then LayerNorm over NL scalars
  float s0 = 1.f / (1.f + __expf(-lambdas[0]));
  float s1 = 1.f / (1.f + __expf(-lambdas[1]));
  float s2 = 1.f / (1.f + __expf(-lambdas[2]));
  float mean = (s0 + s1 + s2) * (1.f / 3.f);
  float var  = ((s0 - mean) * (s0 - mean) + (s1 - mean) * (s1 - mean) + (s2 - mean) * (s2 - mean)) * (1.f / 3.f);
  float li   = (layer == 0) ? s0 : (layer == 1) ? s1 : s2;
  float lam  = (li - mean) * rsqrtf(var + 1e-5f);
  float* trow = total + (size_t)row * EE_DIM;
  for (int e = threadIdx.x; e < EE_DIM; e += 256) {
    float v = a[e] * inv * ln_w[e] * lam;
    if (first) trow[e] = v;
    else       trow[e] += v;
  }
}

// ---------------- final RMS(total + alpha*x) -> bf16 ----------------
__global__ __launch_bounds__(256) void final_rms_kernel(const float* __restrict__ total,
                                                        const float* __restrict__ x,
                                                        const float* __restrict__ alpha,
                                                        const float* __restrict__ fw,
                                                        bf16* __restrict__ y) {
  int row = blockIdx.x;
  const float* trow = total + (size_t)row * EE_DIM;
  const float* xrow = x + (size_t)row * EE_DIM;
  float a = alpha[0];
  float ss = 0.f;
  for (int e = threadIdx.x; e < EE_DIM; e += 256) {
    float v = trow[e] + a * xrow[e];
    ss += v * v;
  }
  __shared__ float red[256];
  red[threadIdx.x] = ss;
  __syncthreads();
  for (int w = 128; w > 0; w >>= 1) {
    if (threadIdx.x < w) red[threadIdx.x] += red[threadIdx.x + w];
    __syncthreads();
  }
  float inv = rsqrtf(red[0] / (float)EE_DIM + 1e-5f);
  bf16* yrow = y + (size_t)row * EE_DIM;
  for (int e = threadIdx.x; e < EE_DIM; e += 256)
    yrow[e] = (bf16)((trow[e] + a * xrow[e]) * inv * fw[e]);
}

// ---------------- host-side orchestration ----------------
extern "C" void kernel_launch(void* const* d_in, const int* in_sizes, int n_in,
                              void* d_out, int out_size, void* d_ws, size_t ws_size,
                              hipStream_t stream) {
  (void)in_sizes; (void)n_in; (void)out_size; (void)ws_size;
  const float* x      = (const float*)d_in[0];
  const float* cosb   = (const float*)d_in[1];
  const float* sinb   = (const float*)d_in[2];
  const float* wq     = (const float*)d_in[3];
  const float* wk     = (const float*)d_in[4];
  const float* wv     = (const float*)d_in[5];
  const float* ln_w   = (const float*)d_in[6];
  const float* lamb   = (const float*)d_in[7];
  const float* w_out  = (const float*)d_in[8];
  const float* flnw   = (const float*)d_in[9];
  const float* alpha  = (const float*)d_in[10];
  float* out = (float*)d_out;

  const size_t NTE = (size_t)BB * TT * EE_DIM;    // 4,194,304
  const size_t EE2 = (size_t)EE_DIM * EE_DIM;     // 4,194,304
  const size_t LMAX = (size_t)TT * 4;             // 4096

  char* ws = (char*)d_ws;
  size_t off = 0;
  auto take = [&](size_t bytes) -> char* {
    char* p = ws + off;
    off += (bytes + 255) & ~(size_t)255;
    return p;
  };
  bf16* xb    = (bf16*)take(NTE * 2);
  bf16* wqb   = (bf16*)take(NL * EE2 * 2);
  bf16* wkb   = (bf16*)take(NL * EE2 * 2);
  bf16* wvb   = (bf16*)take(NL * EE2 * 2);
  bf16* wob   = (bf16*)take(EE2 * 2);
  bf16* qb    = (bf16*)take(NTE * 2);
  bf16* qrope = (bf16*)take(NTE * 2);
  bf16* kraw  = (bf16*)take(NL * NTE * 2);
  bf16* vraw  = (bf16*)take(NL * NTE * 2);
  bf16* kcat  = (bf16*)take((size_t)BB * LMAX * EE_DIM * 2);
  bf16* vt    = (bf16*)take((size_t)BB * LMAX * EE_DIM * 2);
  float* attn = (float*)take(NTE * 4);
  float* total= (float*)take(NTE * 4);
  bf16* ybf   = (bf16*)take(NTE * 2);

  // casts
  cast_f32_bf16_kernel<<<2048, 256, 0, stream>>>(x, xb, NTE);
  cast_f32_bf16_kernel<<<4096, 256, 0, stream>>>(wq, wqb, NL * EE2);
  cast_f32_bf16_kernel<<<4096, 256, 0, stream>>>(wk, wkb, NL * EE2);
  cast_f32_bf16_kernel<<<4096, 256, 0, stream>>>(wv, wvb, NL * EE2);
  cast_f32_bf16_kernel<<<2048, 256, 0, stream>>>(w_out, wob, EE2);

  const int M = BB * TT;  // 2048
  dim3 ggrid(EE_DIM / 128, M / 128);  // (16,16)

  const int Ls[NL]      = {TT, 2 * TT, 4 * TT};
  const int segs[NL][4] = {{0, 0, 0, 0}, {1, 0, 0, 0}, {2, 0, 1, 0}};

  for (int i = 0; i < NL; ++i) {
    // q/k/v projections (bf16 out)
    gemm_wmma_kernel<bf16><<<ggrid, 256, 0, stream>>>(xb, wqb + (size_t)i * EE2, qb, M, EE_DIM, EE_DIM);
    gemm_wmma_kernel<bf16><<<ggrid, 256, 0, stream>>>(xb, wkb + (size_t)i * EE2, kraw + (size_t)i * NTE, M, EE_DIM, EE_DIM);
    gemm_wmma_kernel<bf16><<<ggrid, 256, 0, stream>>>(xb, wvb + (size_t)i * EE2, vraw + (size_t)i * NTE, M, EE_DIM, EE_DIM);

    // rope q (scaled)
    {
      size_t n = (size_t)BB * TT * HH * 64;
      rope_q_kernel<<<(unsigned)((n + 255) / 256), 256, 0, stream>>>(qb, cosb, sinb, qrope);
    }
    // build concat K (re-roped at concat positions) and V^T
    {
      int L = Ls[i];
      size_t n = (size_t)BB * L * HH * 64;
      build_kv_kernel<<<(unsigned)((n + 255) / 256), 256, 0, stream>>>(
          kraw, vraw, cosb, sinb, kcat, vt, L, segs[i][0], segs[i][1], segs[i][2], segs[i][3]);
    }
    // flash attention
    {
      dim3 fg(TT / 64, BB * HH);  // (16, 32)
      flash_attn_kernel<<<fg, 128, 0, stream>>>(qrope, kcat, vt, attn, Ls[i]);
    }
    // rms norm + lambda accumulate
    rms_accum_kernel<<<M, 256, 0, stream>>>(attn, ln_w + (size_t)i * EE_DIM, lamb, i, total, i == 0 ? 1 : 0);
  }

  // final rms + output projection (f32 out)
  final_rms_kernel<<<M, 256, 0, stream>>>(total, x, alpha, flnw, ybf);
  gemm_wmma_kernel<float><<<ggrid, 256, 0, stream>>>(ybf, wob, out, M, EE_DIM, EE_DIM);
}
